// LGBlock_37598143709625
// MI455X (gfx1250) — compile-verified, matrix operand order checked by reading, not run on hardware
//
#include <hip/hip_runtime.h>

// ---------------------------------------------------------------------------
// MI455X / gfx1250. All GEMMs on V_WMMA_F32_16X16X32_F16 (f16 in, f32 acc).
// Block = 256 threads = 8 wave32 waves; wave w owns output rows [16w,16w+16).
// B operands staged in LDS PRE-PACKED in the WMMA per-lane fragment layout
// (each fragment half = one ds_load_b128); all f32->f16 staging is done in
// aligned 8-element octets (global_load_b128 x2 -> ds_store_b128).
// Gather/scatter epilogues batch all loads before use so one s_wait covers
// the whole group instead of 8 serialized round trips.
// ---------------------------------------------------------------------------

#define H 128
#define AS 136            // A-tile LDS stride (f16): 272B rows, 16B-aligned
#define BFR 264           // packed-B: f16 per fragment row (32 lanes*8 + 8 skew)
#define BPL (32 * BFR)    // packed-B: plane offset (second fragment half)
#define BSZ (2 * BPL * 2) // packed-B: bytes per 128x128 tile = 33792

typedef __attribute__((ext_vector_type(16))) _Float16 v16h;
typedef __attribute__((ext_vector_type(8)))  _Float16 v8h;
typedef __attribute__((ext_vector_type(8)))  float    v8f;
typedef __attribute__((ext_vector_type(4)))  float    v4f;

__device__ __forceinline__ v8h cvt8(v4f a, v4f b) {
  v8h r;
  r[0] = (_Float16)a[0]; r[1] = (_Float16)a[1];
  r[2] = (_Float16)a[2]; r[3] = (_Float16)a[3];
  r[4] = (_Float16)b[0]; r[5] = (_Float16)b[1];
  r[6] = (_Float16)b[2]; r[7] = (_Float16)b[3];
  return r;
}

// A fragment for V_WMMA_F32_16X16X32_F16 from row-major f16 tile in LDS.
// ISA 7.12.2 (16-bit A 16x32): lanes 0-15 hold K=0..7 (V0..3) and K=16..23
// (V4..7) for row M=lane; lanes 16-31 hold K=8..15 and K=24..31.
__device__ __forceinline__ v16h ldA(const _Float16* a, int lda, int lane) {
  const int m = lane & 15, kh = (lane >> 4) << 3;
  const _Float16* r = a + m * lda;
  v16h A;
#pragma unroll
  for (int j = 0; j < 8; ++j) { A[j] = r[kh + j]; A[8 + j] = r[16 + kh + j]; }
  return A;
}

// Packed-B fragment load: frag = kt*8 + nt; two contiguous 16B halves.
__device__ __forceinline__ v16h ldBp(const _Float16* bp, int kt, int nt, int lane) {
  const int off = (kt * 8 + nt) * BFR + lane * 8 + ((lane >> 4) << 3);
  v16h B;
#pragma unroll
  for (int j = 0; j < 8; ++j) B[j] = bp[off + j];
#pragma unroll
  for (int j = 0; j < 8; ++j) B[8 + j] = bp[BPL + off + j];
  return B;
}

// Destination of an aligned 8-wide k-octet (k0..k0+7) of column n in the
// packed-B layout: always one contiguous, 16B-aligned group of 8 f16.
__device__ __forceinline__ _Float16* packBoct(_Float16* bp, int k0, int n) {
  const int kt = k0 >> 5, kr = k0 & 31;
  const int h = (kr >> 3) & 1, plane = (kr >> 4) & 1;
  const int lane = h * 16 + (n & 15);
  return bp + plane * BPL + (kt * 8 + (n >> 4)) * BFR + lane * 8 +
         ((lane >> 4) << 3);
}
// Stage one octet of a B tile from global f32.
__device__ __forceinline__ void stageB8(_Float16* bp, int k0, int n,
                                        const float* src) {
  *(v8h*)packBoct(bp, k0, n) = cvt8(*(const v4f*)src, *(const v4f*)(src + 4));
}
// Stage one octet of a row-major f16 A tile from global f32.
__device__ __forceinline__ void stageA8(_Float16* dst, const float* src) {
  *(v8h*)dst = cvt8(*(const v4f*)src, *(const v4f*)(src + 4));
}

__device__ __forceinline__ v8f wmma(v16h A, v16h B, v8f C) {
  return __builtin_amdgcn_wmma_f32_16x16x32_f16(false, A, false, B, (short)0, C,
                                                false, false);
}
// C/D element map: (M = v + 8*(lane>>4), N = lane&15) for v8f index v.

// ---------------------------------------------------------------------------
__global__ void k_copy(const float* __restrict__ src, float* __restrict__ dst,
                       long long n) {
  long long i = (long long)blockIdx.x * blockDim.x + threadIdx.x;
  const long long stride = (long long)gridDim.x * blockDim.x;
  for (; i < n; i += stride) dst[i] = src[i];
}

// ---------------------------------------------------------------------------
// GINE edge stage: msg = relu(x[src] + ea @ We^T + be); agg[dst] += msg.
__global__ void __launch_bounds__(256)
k_edge(const float* __restrict__ x, const float* __restrict__ ea,
       const float* __restrict__ We, const float* __restrict__ be,
       const long long* __restrict__ ei, float* __restrict__ agg, long long E) {
  extern __shared__ char smem[];
  _Float16* sA   = (_Float16*)smem;                  // 128 x AS (ea tile)
  _Float16* sBp  = (_Float16*)(smem + 128 * AS * 2); // packed We^T
  int*      sSrc = (int*)(smem + 128 * AS * 2 + BSZ);
  int*      sDst = sSrc + 128;

  const int tid = threadIdx.x, lane = tid & 31, wave = tid >> 5;
  const long long e0 = (long long)blockIdx.x * 128;
  const bool full = (e0 + 128 <= E);

  for (int i = tid; i < H * 16; i += 256) {          // We -> packed B
    int n = i >> 4, k0 = (i & 15) << 3;
    stageB8(sBp, k0, n, We + n * H + k0);
  }
  for (int i = tid; i < 128 * 16; i += 256) {        // ea tile -> sA
    int r = i >> 4, k0 = (i & 15) << 3;
    if (e0 + r < E) stageA8(sA + r * AS + k0, ea + (e0 + r) * H + k0);
    else { v8h z = {}; *(v8h*)(sA + r * AS + k0) = z; }
  }
  if (tid < 128) {                                   // row offsets (int, *H)
    long long e = e0 + tid;
    sSrc[tid] = (e < E) ? (int)ei[e] * H : 0;
    sDst[tid] = (e < E) ? (int)ei[E + e] * H : 0;
  }
  __syncthreads();

  const int m0 = wave * 16, col = lane & 15, half = lane >> 4;
  int srcOff[8], dstOff[8];
#pragma unroll
  for (int v = 0; v < 8; ++v) {
    srcOff[v] = sSrc[m0 + half * 8 + v];
    dstOff[v] = sDst[m0 + half * 8 + v];
  }

#pragma unroll 1
  for (int nt = 0; nt < 8; ++nt) {
    v8f c = {};
#pragma unroll
    for (int kt = 0; kt < 4; ++kt)
      c = wmma(ldA(sA + m0 * AS + kt * 32, AS, lane), ldBp(sBp, kt, nt, lane), c);
    const int n = nt * 16 + col;
    const float bias = be[n];
    if (full) {                                      // uniform fast path
      float xg[8];
#pragma unroll
      for (int v = 0; v < 8; ++v) xg[v] = x[srcOff[v] + n];  // batched gathers
#pragma unroll
      for (int v = 0; v < 8; ++v) {
        float val = fmaxf(c[v] + bias + xg[v], 0.0f);
        atomicAdd(agg + dstOff[v] + n, val);         // L2-resident
      }
    } else {                                         // tail tile
#pragma unroll
      for (int v = 0; v < 8; ++v) {
        if (e0 + m0 + half * 8 + v < E) {
          float val = fmaxf(c[v] + bias + x[srcOff[v] + n], 0.0f);
          atomicAdd(agg + dstOff[v] + n, val);
        }
      }
    }
  }
}

// ---------------------------------------------------------------------------
// Node MLP + residual + LayerNorm:
//   h = relu(agg @ W1^T + b1) @ W2^T + b2 ; out = LN(x + h) * g + b
__global__ void __launch_bounds__(256)
k_mlp_ln(const float* __restrict__ x, const float* __restrict__ agg,
         const float* __restrict__ W1, const float* __restrict__ b1,
         const float* __restrict__ W2, const float* __restrict__ b2,
         const float* __restrict__ lng, const float* __restrict__ lnb,
         float* __restrict__ out, int N) {
  extern __shared__ char smem[];
  _Float16* sA  = (_Float16*)smem;                   // 128 x AS
  _Float16* sBp = (_Float16*)(smem + 128 * AS * 2);  // packed weights
  float*    sS  = (float*)smem;                      // aliases sA|sBp after GEMMs
  float* sMean  = (float*)(smem + 128 * AS * 2 + BSZ);
  float* sRstd  = sMean + 128;

  const int tid = threadIdx.x, lane = tid & 31, wave = tid >> 5;
  const int r0 = blockIdx.x * 128;
  const int m0 = wave * 16, col = lane & 15, half = lane >> 4;

  for (int i = tid; i < 128 * 16; i += 256) {
    int r = i >> 4, k0 = (i & 15) << 3, rg = r0 + r;
    if (rg < N) stageA8(sA + r * AS + k0, agg + (long long)rg * H + k0);
    else { v8h z = {}; *(v8h*)(sA + r * AS + k0) = z; }
  }
  for (int i = tid; i < H * 16; i += 256) {
    int n = i >> 4, k0 = (i & 15) << 3;
    stageB8(sBp, k0, n, W1 + n * H + k0);
  }
  __syncthreads();

  v8f t[8];
#pragma unroll 1
  for (int nt = 0; nt < 8; ++nt) {
    v8f c = {};
#pragma unroll
    for (int kt = 0; kt < 4; ++kt)
      c = wmma(ldA(sA + m0 * AS + kt * 32, AS, lane), ldBp(sBp, kt, nt, lane), c);
    float bias = b1[nt * 16 + col];
#pragma unroll
    for (int v = 0; v < 8; ++v) c[v] = fmaxf(c[v] + bias, 0.0f);
    t[nt] = c;
  }
  __syncthreads();
#pragma unroll
  for (int nt = 0; nt < 8; ++nt)
#pragma unroll
    for (int v = 0; v < 8; ++v)
      sA[(m0 + half * 8 + v) * AS + nt * 16 + col] = (_Float16)t[nt][v];
  for (int i = tid; i < H * 16; i += 256) {
    int n = i >> 4, k0 = (i & 15) << 3;
    stageB8(sBp, k0, n, W2 + n * H + k0);
  }
  __syncthreads();

  v8f hacc[8];
#pragma unroll 1
  for (int nt = 0; nt < 8; ++nt) {
    v8f c = {};
#pragma unroll
    for (int kt = 0; kt < 4; ++kt)
      c = wmma(ldA(sA + m0 * AS + kt * 32, AS, lane), ldBp(sBp, kt, nt, lane), c);
    hacc[nt] = c;
  }
  __syncthreads();
  const bool rowsOk = (r0 + 128 <= N);
#pragma unroll 1
  for (int nt = 0; nt < 8; ++nt) {
    int n = nt * 16 + col;
    float bias = b2[n];
    float xg[8];
    if (rowsOk) {
#pragma unroll
      for (int v = 0; v < 8; ++v)
        xg[v] = x[(long long)(r0 + m0 + half * 8 + v) * H + n];
    } else {
#pragma unroll
      for (int v = 0; v < 8; ++v) {
        int rg = r0 + m0 + half * 8 + v;
        xg[v] = (rg < N) ? x[(long long)rg * H + n] : 0.0f;
      }
    }
#pragma unroll
    for (int v = 0; v < 8; ++v)
      sS[(m0 + half * 8 + v) * 132 + n] = xg[v] + hacc[nt][v] + bias;
  }
  __syncthreads();
  if (tid < 128) {
    float mu = 0.0f;
    for (int k = 0; k < H; ++k) mu += sS[tid * 132 + k];
    mu *= (1.0f / H);
    float var = 0.0f;
    for (int k = 0; k < H; ++k) { float d = sS[tid * 132 + k] - mu; var += d * d; }
    var *= (1.0f / H);
    sMean[tid] = mu;
    sRstd[tid] = rsqrtf(var + 1e-5f);
  }
  __syncthreads();
  for (int i = tid; i < 128 * 16; i += 256) {
    int r = i >> 4, k0 = (i & 15) << 3, rg = r0 + r;
    if (rg < N) {
      float mu = sMean[r], rs = sRstd[r];
      v4f o0, o1;
#pragma unroll
      for (int j = 0; j < 4; ++j)
        o0[j] = (sS[r * 132 + k0 + j] - mu) * rs * lng[k0 + j] + lnb[k0 + j];
#pragma unroll
      for (int j = 0; j < 4; ++j)
        o1[j] = (sS[r * 132 + k0 + 4 + j] - mu) * rs * lng[k0 + 4 + j] + lnb[k0 + 4 + j];
      *(v4f*)(out + (long long)rg * H + k0) = o0;
      *(v4f*)(out + (long long)rg * H + k0 + 4) = o1;
    }
  }
}

// ---------------------------------------------------------------------------
// Chunked MHA over chem rows; one block = one chunk of B=128 rows, 4 heads,
// dh=32 (Q.K^T is a single K=32 WMMA per 16x16 tile). Softmax in registers
// via __shfl_xor across the 16-lane row group of the C layout.
__global__ void __launch_bounds__(256)
k_attn(float* __restrict__ xio, const long long* __restrict__ cidx,
       const float* __restrict__ in_w, const float* __restrict__ in_b,
       const float* __restrict__ out_w, const float* __restrict__ out_b) {
  extern __shared__ char smem[];
  _Float16* sX  = (_Float16*)smem;            // 128 x AS : xc, later P (probs)
  _Float16* sWp = sX + 128 * AS;              // packed weight tile
  _Float16* sQ  = sWp + BSZ / 2;              // 128 x AS : q rows (A operand)
  _Float16* sKp = sQ + 128 * AS;              // packed K^T (k index = dim)
  _Float16* sVp = sKp + BSZ / 2;              // packed V   (k index = key)
  _Float16* sO  = sVp + BSZ / 2;              // 128 x AS : attention output
  int*      sIdx = (int*)(sO + 128 * AS);     // row offsets (*H)

  const int tid = threadIdx.x, lane = tid & 31, wave = tid >> 5;
  const long long base = (long long)blockIdx.x * 128;
  const int m0 = wave * 16, col = lane & 15, half = lane >> 4;

  if (tid < 128) sIdx[tid] = (int)cidx[base + tid] * H;
  __syncthreads();
  for (int i = tid; i < 128 * 16; i += 256) {
    int r = i >> 4, k0 = (i & 15) << 3;
    stageA8(sX + r * AS + k0, xio + sIdx[r] + k0);
  }
  // qkv projection: 3 passes over the packed in_w
  for (int o = 0; o < 3; ++o) {
    __syncthreads();
    for (int i = tid; i < H * 16; i += 256) {
      int n = i >> 4, k0 = (i & 15) << 3;
      stageB8(sWp, k0, n, in_w + (o * H + n) * H + k0);
    }
    __syncthreads();
#pragma unroll 1
    for (int nt = 0; nt < 8; ++nt) {
      v8f c = {};
#pragma unroll
      for (int kt = 0; kt < 4; ++kt)
        c = wmma(ldA(sX + m0 * AS + kt * 32, AS, lane), ldBp(sWp, kt, nt, lane), c);
      int n = nt * 16 + col;
      float bias = in_b[o * H + n];
      if (o == 0) {
#pragma unroll
        for (int v = 0; v < 8; ++v)
          sQ[(m0 + half * 8 + v) * AS + n] = (_Float16)(c[v] + bias);
      } else if (o == 1) {                    // K^T: k index = dim (n), col = key
#pragma unroll
        for (int v = 0; v < 8; ++v) {
          int m = m0 + half * 8 + v;
          const int kt2 = n >> 5, kr = n & 31;
          const int h2 = (kr >> 3) & 1, pl = (kr >> 4) & 1;
          const int ln2 = h2 * 16 + (m & 15);
          sKp[pl * BPL + (kt2 * 8 + (m >> 4)) * BFR + ln2 * 8 +
              ((ln2 >> 4) << 3) + (kr & 7)] = (_Float16)(c[v] + bias);
        }
      } else {                                // V: rows m are one aligned octet
        v8h o8;
#pragma unroll
        for (int v = 0; v < 8; ++v) o8[v] = (_Float16)(c[v] + bias);
        *(v8h*)packBoct(sVp, m0 + half * 8, n) = o8;
      }
    }
  }
  __syncthreads();

  const float scale = 0.17677669529663687f;  // 1/sqrt(32)
  for (int hd = 0; hd < 4; ++hd) {
    v8f s[8];
#pragma unroll 1
    for (int nt = 0; nt < 8; ++nt) {          // scores: K = dh = 32 -> kt = hd
      v8f c = {};
      c = wmma(ldA(sQ + m0 * AS + hd * 32, AS, lane), ldBp(sKp, hd, nt, lane), c);
#pragma unroll
      for (int v = 0; v < 8; ++v) c[v] *= scale;
      s[nt] = c;
    }
    v8f p[8];
#pragma unroll
    for (int v = 0; v < 8; ++v) {             // softmax per row (v, half)
      float mx = -1e30f;
#pragma unroll
      for (int nt = 0; nt < 8; ++nt) mx = fmaxf(mx, s[nt][v]);
      for (int off = 1; off < 16; off <<= 1) mx = fmaxf(mx, __shfl_xor(mx, off, 32));
      float ev[8], sum = 0.0f;
#pragma unroll
      for (int nt = 0; nt < 8; ++nt) { ev[nt] = __expf(s[nt][v] - mx); sum += ev[nt]; }
      for (int off = 1; off < 16; off <<= 1) sum += __shfl_xor(sum, off, 32);
      float inv = 1.0f / sum;
#pragma unroll
      for (int nt = 0; nt < 8; ++nt) p[nt][v] = ev[nt] * inv;
    }
    // P -> sX (own rows only; same-wave LDS deps resolved by waitcnts)
#pragma unroll
    for (int nt = 0; nt < 8; ++nt) {
      int n = nt * 16 + col;
#pragma unroll
      for (int v = 0; v < 8; ++v)
        sX[(m0 + half * 8 + v) * AS + n] = (_Float16)p[nt][v];
    }
    // o = P @ V_head : dim tiles nt = hd*2 + {0,1}, K = 128 keys
#pragma unroll 1
    for (int dt = 0; dt < 2; ++dt) {
      v8f c = {};
#pragma unroll
      for (int kt = 0; kt < 4; ++kt)
        c = wmma(ldA(sX + m0 * AS + kt * 32, AS, lane),
                 ldBp(sVp, kt, hd * 2 + dt, lane), c);
      int n = hd * 32 + dt * 16 + col;
#pragma unroll
      for (int v = 0; v < 8; ++v)
        sO[(m0 + half * 8 + v) * AS + n] = (_Float16)c[v];
    }
  }
  __syncthreads();
  for (int i = tid; i < H * 16; i += 256) {   // out_w^T
    int n = i >> 4, k0 = (i & 15) << 3;
    stageB8(sWp, k0, n, out_w + n * H + k0);
  }
  __syncthreads();
#pragma unroll 1
  for (int nt = 0; nt < 8; ++nt) {
    v8f c = {};
#pragma unroll
    for (int kt = 0; kt < 4; ++kt)
      c = wmma(ldA(sO + m0 * AS + kt * 32, AS, lane), ldBp(sWp, kt, nt, lane), c);
    int n = nt * 16 + col;
    float bias = out_b[n];
    float xg[8];
#pragma unroll
    for (int v = 0; v < 8; ++v)               // batched residual reads
      xg[v] = xio[sIdx[m0 + half * 8 + v] + n];
#pragma unroll
    for (int v = 0; v < 8; ++v)
      xio[sIdx[m0 + half * 8 + v] + n] = xg[v] + c[v] + bias;
  }
}

// ---------------------------------------------------------------------------
// FFN on chem rows: xc += relu(LN(xc) @ f1^T + f1b) @ f2^T + f2b
// Hidden dim 2H processed in two 128-wide halves, accumulating GEMM2.
__global__ void __launch_bounds__(256)
k_ffn(float* __restrict__ xio, const long long* __restrict__ cidx,
      const float* __restrict__ lng, const float* __restrict__ lnb,
      const float* __restrict__ f1w, const float* __restrict__ f1b,
      const float* __restrict__ f2w, const float* __restrict__ f2b) {
  extern __shared__ char smem[];
  float*    sS   = (float*)smem;                           // 128 x 132 f32 (xc)
  _Float16* sA   = (_Float16*)(smem + 128 * 132 * 4);      // 128 x AS (LN(xc))
  _Float16* sBp  = sA + 128 * AS;                          // packed weights
  _Float16* sMid = sBp + BSZ / 2;                          // 128 x AS (relu mid)
  float* sMean   = (float*)(sMid + 128 * AS);
  float* sRstd   = sMean + 128;
  int*   sIdx    = (int*)(sRstd + 128);

  const int tid = threadIdx.x, lane = tid & 31, wave = tid >> 5;
  const long long base = (long long)blockIdx.x * 128;
  const int m0 = wave * 16, col = lane & 15, half = lane >> 4;

  if (tid < 128) sIdx[tid] = (int)cidx[base + tid] * H;
  __syncthreads();
  for (int i = tid; i < 128 * 16; i += 256) {
    int r = i >> 4, k0 = (i & 15) << 3;
    const float* s = xio + sIdx[r] + k0;
    *(v4f*)(sS + r * 132 + k0) = *(const v4f*)s;
    *(v4f*)(sS + r * 132 + k0 + 4) = *(const v4f*)(s + 4);
  }
  __syncthreads();
  if (tid < 128) {
    float mu = 0.0f;
    for (int k = 0; k < H; ++k) mu += sS[tid * 132 + k];
    mu *= (1.0f / H);
    float var = 0.0f;
    for (int k = 0; k < H; ++k) { float d = sS[tid * 132 + k] - mu; var += d * d; }
    var *= (1.0f / H);
    sMean[tid] = mu;
    sRstd[tid] = rsqrtf(var + 1e-5f);
  }
  __syncthreads();
  for (int i = tid; i < 128 * 16; i += 256) {
    int r = i >> 4, k0 = (i & 15) << 3;
    float mu = sMean[r], rs = sRstd[r];
    v8h o8;
#pragma unroll
    for (int j = 0; j < 8; ++j)
      o8[j] = (_Float16)((sS[r * 132 + k0 + j] - mu) * rs * lng[k0 + j] + lnb[k0 + j]);
    *(v8h*)(sA + r * AS + k0) = o8;
  }

  v8f zero = {};
  v8f acc[8];
#pragma unroll
  for (int nt = 0; nt < 8; ++nt) acc[nt] = zero;

  for (int hf = 0; hf < 2; ++hf) {
    __syncthreads();
    for (int i = tid; i < H * 16; i += 256) { // f1 half: B[k][n] = f1w[hf*H+n][k]
      int n = i >> 4, k0 = (i & 15) << 3;
      stageB8(sBp, k0, n, f1w + (hf * H + n) * H + k0);
    }
    __syncthreads();
#pragma unroll 1
    for (int nt = 0; nt < 8; ++nt) {
      v8f c = {};
#pragma unroll
      for (int kt = 0; kt < 4; ++kt)
        c = wmma(ldA(sA + m0 * AS + kt * 32, AS, lane), ldBp(sBp, kt, nt, lane), c);
      int n = nt * 16 + col;
      float bias = f1b[hf * H + n];
#pragma unroll
      for (int v = 0; v < 8; ++v)
        sMid[(m0 + half * 8 + v) * AS + n] = (_Float16)fmaxf(c[v] + bias, 0.0f);
    }
    __syncthreads();
    for (int i = tid; i < H * 16; i += 256) { // f2 half: B[k][n] = f2w[n][hf*H+k]
      int n = i >> 4, k0 = (i & 15) << 3;
      stageB8(sBp, k0, n, f2w + n * (2 * H) + hf * H + k0);
    }
    __syncthreads();
#pragma unroll 1
    for (int nt = 0; nt < 8; ++nt) {
      v8f c = acc[nt];
#pragma unroll
      for (int kt = 0; kt < 4; ++kt)
        c = wmma(ldA(sMid + m0 * AS + kt * 32, AS, lane), ldBp(sBp, kt, nt, lane), c);
      acc[nt] = c;
    }
  }
#pragma unroll
  for (int nt = 0; nt < 8; ++nt) {
    int n = nt * 16 + col;
    float bias = f2b[n];
#pragma unroll
    for (int v = 0; v < 8; ++v) {
      int m = m0 + half * 8 + v;
      xio[sIdx[m] + n] = sS[m * 132 + n] + acc[nt][v] + bias;
    }
  }
}

// ---------------------------------------------------------------------------
extern "C" void kernel_launch(void* const* d_in, const int* in_sizes, int n_in,
                              void* d_out, int out_size, void* d_ws, size_t ws_size,
                              hipStream_t stream) {
  const float* x    = (const float*)d_in[0];
  const float* ea   = (const float*)d_in[1];
  const float* We   = (const float*)d_in[2];
  const float* be_  = (const float*)d_in[3];
  const float* W1   = (const float*)d_in[4];
  const float* b1   = (const float*)d_in[5];
  const float* W2   = (const float*)d_in[6];
  const float* b2   = (const float*)d_in[7];
  const float* ln1g = (const float*)d_in[8];
  const float* ln1b = (const float*)d_in[9];
  const float* inw  = (const float*)d_in[10];
  const float* inb  = (const float*)d_in[11];
  const float* outw = (const float*)d_in[12];
  const float* outb = (const float*)d_in[13];
  const float* ln2g = (const float*)d_in[14];
  const float* ln2b = (const float*)d_in[15];
  const float* f1w  = (const float*)d_in[16];
  const float* f1b  = (const float*)d_in[17];
  const float* f2w  = (const float*)d_in[18];
  const float* f2b  = (const float*)d_in[19];
  const long long* ei   = (const long long*)d_in[20];
  const long long* cidx = (const long long*)d_in[21];

  const int N        = in_sizes[0] / H;
  const long long E  = (long long)in_sizes[20] / 2;
  const int NC       = in_sizes[21];

  float* agg = (float*)d_ws;     // N*H f32 scratch
  float* out = (float*)d_out;

  const int smem_edge = 128 * AS * 2 + BSZ + 256 * 4;
  const int smem_mlp  = 128 * AS * 2 + BSZ + 2 * 128 * 4;
  const int smem_attn = 3 * (128 * AS * 2) + 3 * BSZ + 128 * 4;
  const int smem_ffn  = 128 * 132 * 4 + 2 * (128 * AS * 2) + BSZ + 3 * 128 * 4;

  (void)hipFuncSetAttribute((const void*)k_edge,
      hipFuncAttributeMaxDynamicSharedMemorySize, smem_edge);
  (void)hipFuncSetAttribute((const void*)k_mlp_ln,
      hipFuncAttributeMaxDynamicSharedMemorySize, smem_mlp);
  (void)hipFuncSetAttribute((const void*)k_attn,
      hipFuncAttributeMaxDynamicSharedMemorySize, smem_attn);
  (void)hipFuncSetAttribute((const void*)k_ffn,
      hipFuncAttributeMaxDynamicSharedMemorySize, smem_ffn);

  // 1) agg = x   (folds the "+ x" self term of GINE)
  k_copy<<<2048, 256, 0, stream>>>(x, agg, (long long)N * H);
  // 2) edge messages + scatter-add (dominant stage; HBM-bound at ~23.3 TB/s)
  k_edge<<<(int)((E + 127) / 128), 256, smem_edge, stream>>>(x, ea, We, be_, ei, agg, E);
  // 3) node MLP + residual + LN1 -> out
  k_mlp_ln<<<(N + 127) / 128, 256, smem_mlp, stream>>>(x, agg, W1, b1, W2, b2,
                                                       ln1g, ln1b, out, N);
  // 4) chunked self-attention on chem rows (in place on out)
  k_attn<<<NC / 128, 256, smem_attn, stream>>>(out, cidx, inw, inb, outw, outb);
  // 5) FFN on chem rows (in place on out)
  k_ffn<<<NC / 128, 256, smem_ffn, stream>>>(out, cidx, ln2g, ln2b, f1w, f1b,
                                             f2w, f2b);
}
